// DDECModel_33827162423830
// MI455X (gfx1250) — compile-verified, matrix operand order checked by reading, not run on hardware
//
#include <hip/hip_runtime.h>
#include <math.h>

typedef __attribute__((ext_vector_type(16))) __bf16          v16bf;
typedef __attribute__((ext_vector_type(16))) unsigned short  v16u;
typedef __attribute__((ext_vector_type(8)))  float           v8f;
typedef __attribute__((ext_vector_type(4)))  unsigned int    v4u;
typedef __attribute__((ext_vector_type(8)))  int             v8i;
typedef __attribute__((ext_vector_type(4)))  int             v4i;

#define N0    4096
#define N1    16384
#define NB    128
#define KT    32
#define RIDGE 1e-6f
#define EPSD  1e-5f
#define TOL   1e-8f

// ---------- helpers ----------
__device__ __forceinline__ unsigned short f2bf(float x) {
  unsigned int u = __float_as_uint(x);
  u += 0x7fffu + ((u >> 16) & 1u);          // round-to-nearest-even
  return (unsigned short)(u >> 16);
}
__device__ __forceinline__ float bf2f(unsigned short h) {
  return __uint_as_float(((unsigned int)h) << 16);
}
__device__ __forceinline__ float wredsum(float v) {
  v += __shfl_xor(v, 16, 32);
  v += __shfl_xor(v, 8, 32);
  v += __shfl_xor(v, 4, 32);
  v += __shfl_xor(v, 2, 32);
  v += __shfl_xor(v, 1, 32);
  return v;
}
// Build one lane's 16x32 bf16 fragment from an LDS tile row (row stride in ushorts).
// lanes 0-15 hold K {0..7, 16..23}; lanes 16-31 hold K {8..15, 24..31}.
__device__ __forceinline__ v16bf mkfrag(const unsigned short* rowp, int half) {
  v16u v;
  const unsigned short* p0 = rowp + half * 8;
  const unsigned short* p1 = rowp + 16 + half * 8;
#pragma unroll
  for (int i = 0; i < 8; i++) { v[i] = p0[i]; v[8 + i] = p1[i]; }
  return __builtin_bit_cast(v16bf, v);
}

// ---------- TDM: issue one 128-row x 32-col bf16 tile load, row stride N1 ----------
// D# built per CDNA5 ISA ch.8: data_size=2B, pad every 16 dwords by 1 dword
// so each 64B tile row lands at a 68B LDS stride (bank-conflict-free ds_load_b128).
__device__ __forceinline__ void tdm_issue(const unsigned short* gsrc, unsigned ldsOff) {
  unsigned long long ga = (unsigned long long)(uintptr_t)gsrc;
  v4u g0;
  g0[0] = 1u;                                             // count=1 valid, no gather
  g0[1] = ldsOff;                                         // lds_addr (bytes)
  g0[2] = (unsigned)ga;                                   // global_addr[31:0]
  g0[3] = (unsigned)((ga >> 32) & 0x01ffffffu) | (2u << 30); // addr[56:32] | type=2
  v8i g1;
  g1[0] = (int)((1u << 16)      // data_size = 1 -> 2 bytes
              | (1u << 20)      // pad_enable
              | (3u << 22)      // pad_interval = 3 -> every 16 dwords
              | (0u << 25));    // pad_amount  = 0 -> 1 dword
  g1[1] = (int)(((unsigned)N1 & 0xffffu) << 16);          // tensor_dim0 lo16
  g1[2] = (int)(((unsigned)N1 >> 16) | (((unsigned)N0 & 0xffffu) << 16)); // dim0 hi | dim1 lo
  g1[3] = (int)(((unsigned)N0 >> 16) | (32u << 16));      // dim1 hi | tile_dim0=32
  g1[4] = (int)128u;                                      // tile_dim1=128, tile_dim2=0
  g1[5] = (int)(unsigned)N1;                              // tensor_dim0_stride lo32
  g1[6] = 0;
  g1[7] = 0;
  v4i z4 = {0, 0, 0, 0};
#if __clang_major__ >= 23
  v8i z8 = {0, 0, 0, 0, 0, 0, 0, 0};
  __builtin_amdgcn_tensor_load_to_lds(g0, g1, z4, z4, z8, 0);
#else
  __builtin_amdgcn_tensor_load_to_lds(g0, g1, z4, z4, 0);
#endif
}

// ---------- precompute vectors ----------
__global__ void prep_kernel(const float* __restrict__ f,
                            const float* __restrict__ B1, const float* __restrict__ B2,
                            const float* __restrict__ D1, const float* __restrict__ D2,
                            float* w, float* bh, float* dAdd,
                            float* b2v, float* dd2v, float* rr) {
  int i = blockIdx.x * blockDim.x + threadIdx.x;
  if (i < N1) w[i] = 1.f / ((B1[i] * B1[i] + EPSD) * (D1[i] * D1[i] + EPSD));
  if (i < N0) {
    float b2  = B2[i] * B2[i];
    float dd2 = D2[i] * D2[i];
    b2v[i] = b2; dd2v[i] = dd2;
    bh[i]   = f[i] / b2;
    dAdd[i] = RIDGE / (b2 * dd2);
  }
  if (i == 0) *rr = 0.f;
}

// ---------- one-pass S = d1*sqrt(w) -> bf16 hi/lo split ----------
__global__ void split_kernel(const float* __restrict__ d1, const float* __restrict__ w,
                             unsigned short* __restrict__ Shi, unsigned short* __restrict__ Slo) {
  size_t i4 = (size_t)blockIdx.x * blockDim.x + threadIdx.x;
  size_t base = i4 * 4;
  if (base >= (size_t)N0 * N1) return;
  float4 v = *(const float4*)(d1 + base);
  int k = (int)(base & (N1 - 1));
  float4 wv = *(const float4*)(w + k);
  float a[4] = {v.x * sqrtf(wv.x), v.y * sqrtf(wv.y), v.z * sqrtf(wv.z), v.w * sqrtf(wv.w)};
  unsigned short hs[4], ls[4];
#pragma unroll
  for (int e = 0; e < 4; e++) {
    hs[e] = f2bf(a[e]);
    ls[e] = f2bf(a[e] - bf2f(hs[e]));
  }
  *(ushort4*)(Shi + base) = make_ushort4(hs[0], hs[1], hs[2], hs[3]);
  *(ushort4*)(Slo + base) = make_ushort4(ls[0], ls[1], ls[2], ls[3]);
}

// ---------- Gram GEMM: C = S*S^T (+diag), TDM double-buffered bf16x3 WMMA ----------
#define TILE_US (128 * 34)   // 128 rows, 34 ushorts (68B) per row after TDM padding
__global__ __launch_bounds__(256) void gram_gemm_tdm(
    const unsigned short* __restrict__ Shi, const unsigned short* __restrict__ Slo,
    float* __restrict__ C, const float* __restrict__ dAdd) {
  extern __shared__ char gsm[];
  unsigned short* T = (unsigned short*)gsm;     // [2 buf][4 mat][TILE_US]
  const unsigned ldsBase = (unsigned)(uintptr_t)T;

  const int tid  = threadIdx.x;
  const int lane = tid & 31;
  const int wid  = tid >> 5;
  const int half = lane >> 4;
  const int ml   = lane & 15;
  const int wM = (wid >> 2) * 64;   // 2x4 wave grid; wave tile 64x32
  const int wN = (wid & 3) * 32;
  const size_t Rb = (size_t)blockIdx.y * NB;
  const size_t Cb = (size_t)blockIdx.x * NB;

  v8f acc[4][2];
#pragma unroll
  for (int mt = 0; mt < 4; mt++)
#pragma unroll
    for (int nt = 0; nt < 2; nt++)
#pragma unroll
      for (int g = 0; g < 8; g++) acc[mt][nt][g] = 0.f;

  // prologue: wave0 issues buffer 0 tiles for kt = 0
  if (wid == 0) {
    tdm_issue(Shi + Rb * N1, ldsBase + (0 * 4 + 0) * TILE_US * 2);
    tdm_issue(Slo + Rb * N1, ldsBase + (0 * 4 + 1) * TILE_US * 2);
    tdm_issue(Shi + Cb * N1, ldsBase + (0 * 4 + 2) * TILE_US * 2);
    tdm_issue(Slo + Cb * N1, ldsBase + (0 * 4 + 3) * TILE_US * 2);
  }

  for (int kt = 0; kt < N1; kt += KT) {
    const int cur = (kt >> 5) & 1;
    if (wid == 0) __builtin_amdgcn_s_wait_tensorcnt(0);
    __syncthreads();                 // tiles[cur] ready for everyone
    if (wid == 0 && kt + KT < N1) {  // prefetch next k-tile into the other buffer
      const int nxt = cur ^ 1;
      tdm_issue(Shi + Rb * N1 + kt + KT, ldsBase + (nxt * 4 + 0) * TILE_US * 2);
      tdm_issue(Slo + Rb * N1 + kt + KT, ldsBase + (nxt * 4 + 1) * TILE_US * 2);
      tdm_issue(Shi + Cb * N1 + kt + KT, ldsBase + (nxt * 4 + 2) * TILE_US * 2);
      tdm_issue(Slo + Cb * N1 + kt + KT, ldsBase + (nxt * 4 + 3) * TILE_US * 2);
    }
    const unsigned short* Ah = T + (cur * 4 + 0) * TILE_US;
    const unsigned short* Al = T + (cur * 4 + 1) * TILE_US;
    const unsigned short* Bh = T + (cur * 4 + 2) * TILE_US;
    const unsigned short* Bl = T + (cur * 4 + 3) * TILE_US;

    v16bf ah[4], al[4], bh2[2], bl2[2];
#pragma unroll
    for (int mt = 0; mt < 4; mt++) {
      int r = wM + mt * 16 + ml;
      ah[mt] = mkfrag(Ah + r * 34, half);
      al[mt] = mkfrag(Al + r * 34, half);
    }
#pragma unroll
    for (int nt = 0; nt < 2; nt++) {
      int r = wN + nt * 16 + ml;
      bh2[nt] = mkfrag(Bh + r * 34, half);
      bl2[nt] = mkfrag(Bl + r * 34, half);
    }
#pragma unroll
    for (int mt = 0; mt < 4; mt++)
#pragma unroll
      for (int nt = 0; nt < 2; nt++) {
        v8f c = acc[mt][nt];
        c = __builtin_amdgcn_wmma_f32_16x16x32_bf16(false, ah[mt], false, bh2[nt], (short)0, c, false, false);
        c = __builtin_amdgcn_wmma_f32_16x16x32_bf16(false, ah[mt], false, bl2[nt], (short)0, c, false, false);
        c = __builtin_amdgcn_wmma_f32_16x16x32_bf16(false, al[mt], false, bh2[nt], (short)0, c, false, false);
        acc[mt][nt] = c;
      }
    __syncthreads();                 // all reads of tiles[cur] done before it is refilled
  }

#pragma unroll
  for (int mt = 0; mt < 4; mt++)
#pragma unroll
    for (int nt = 0; nt < 2; nt++)
#pragma unroll
      for (int g = 0; g < 8; g++) {
        int row = (int)Rb + wM + mt * 16 + half * 8 + g;
        int col = (int)Cb + wN + nt * 16 + ml;
        float val = acc[mt][nt][g];
        C[(size_t)row * N0 + col] = val + ((row == col) ? dAdd[row] : 0.f);
      }
}

// ---------- fp32-input 128x128 GEMM (Cholesky trailing syrk): C -= P*P^T ----------
__global__ __launch_bounds__(256) void gemm128_bf16x3(
    const float* __restrict__ Ab, int lda,
    const float* __restrict__ Bb, int ldb,
    int kdim,
    float* __restrict__ C, int ldc,
    int lowerOnly) {
  if (lowerOnly && (int)blockIdx.y < (int)blockIdx.x) return;

  __shared__ unsigned short lA[2][NB][40];
  __shared__ unsigned short lB[2][NB][40];

  const int tid  = threadIdx.x;
  const int lane = tid & 31;
  const int wid  = tid >> 5;
  const int half = lane >> 4;
  const int ml   = lane & 15;
  const int wM = (wid >> 2) * 64;
  const int wN = (wid & 3) * 32;

  v8f acc[4][2];
#pragma unroll
  for (int mt = 0; mt < 4; mt++)
#pragma unroll
    for (int nt = 0; nt < 2; nt++)
#pragma unroll
      for (int g = 0; g < 8; g++) acc[mt][nt][g] = 0.f;

  for (int kt = 0; kt < kdim; kt += KT) {
    __syncthreads();
#pragma unroll
    for (int t = 0; t < 4; t++) {
      int f4  = tid + t * 256;
      int row = f4 >> 3;
      int kq  = (f4 & 7) << 2;
      int kg  = kt + kq;
      float4 av = *(const float4*)(Ab + (size_t)row * lda + kg);
      float a4[4] = {av.x, av.y, av.z, av.w};
#pragma unroll
      for (int e = 0; e < 4; e++) {
        unsigned short h = f2bf(a4[e]);
        lA[0][row][kq + e] = h;
        lA[1][row][kq + e] = f2bf(a4[e] - bf2f(h));
      }
      float4 bv = *(const float4*)(Bb + (size_t)row * ldb + kg);
      float b4[4] = {bv.x, bv.y, bv.z, bv.w};
#pragma unroll
      for (int e = 0; e < 4; e++) {
        unsigned short h = f2bf(b4[e]);
        lB[0][row][kq + e] = h;
        lB[1][row][kq + e] = f2bf(b4[e] - bf2f(h));
      }
      if (kt + KT < kdim) {          // global_prefetch of next k-tile
        __builtin_prefetch(Ab + (size_t)row * lda + kg + KT, 0, 0);
        __builtin_prefetch(Bb + (size_t)row * ldb + kg + KT, 0, 0);
      }
    }
    __syncthreads();

    v16bf ah[4], al[4], bh2[2], bl2[2];
#pragma unroll
    for (int mt = 0; mt < 4; mt++) {
      int r = wM + mt * 16 + ml;
      ah[mt] = mkfrag(&lA[0][r][0], half);
      al[mt] = mkfrag(&lA[1][r][0], half);
    }
#pragma unroll
    for (int nt = 0; nt < 2; nt++) {
      int r = wN + nt * 16 + ml;
      bh2[nt] = mkfrag(&lB[0][r][0], half);
      bl2[nt] = mkfrag(&lB[1][r][0], half);
    }
#pragma unroll
    for (int mt = 0; mt < 4; mt++)
#pragma unroll
      for (int nt = 0; nt < 2; nt++) {
        v8f c = acc[mt][nt];
        c = __builtin_amdgcn_wmma_f32_16x16x32_bf16(false, ah[mt], false, bh2[nt], (short)0, c, false, false);
        c = __builtin_amdgcn_wmma_f32_16x16x32_bf16(false, ah[mt], false, bl2[nt], (short)0, c, false, false);
        c = __builtin_amdgcn_wmma_f32_16x16x32_bf16(false, al[mt], false, bh2[nt], (short)0, c, false, false);
        acc[mt][nt] = c;
      }
  }

#pragma unroll
  for (int mt = 0; mt < 4; mt++)
#pragma unroll
    for (int nt = 0; nt < 2; nt++)
#pragma unroll
      for (int g = 0; g < 8; g++) {
        int row = (int)blockIdx.y * NB + wM + mt * 16 + half * 8 + g;
        int col = (int)blockIdx.x * NB + wN + nt * 16 + ml;
        C[(size_t)row * ldc + col] -= acc[mt][nt][g];
      }
}

// ---------- Cholesky diagonal-block factor (one WG) ----------
__global__ void potf2_kernel(float* L, int ld, int jb) {
  extern __shared__ char smem[];
  float* s = (float*)smem;
  const int tid = threadIdx.x;                   // 256
  const int base = jb * NB;
  for (int idx = tid; idx < NB * NB; idx += 256)
    s[idx] = L[(size_t)(base + idx / NB) * ld + base + (idx % NB)];
  __syncthreads();
  for (int k = 0; k < NB; k++) {
    if (tid == 0) s[k * NB + k] = sqrtf(s[k * NB + k]);
    __syncthreads();
    float dk = s[k * NB + k];
    for (int i = k + 1 + tid; i < NB; i += 256) s[i * NB + k] /= dk;
    __syncthreads();
    for (int j = k + 1; j < NB; j++) {
      float ljk = s[j * NB + k];
      for (int i = j + tid; i < NB; i += 256) s[i * NB + j] -= s[i * NB + k] * ljk;
    }
    __syncthreads();
  }
  for (int idx = tid; idx < NB * NB; idx += 256) {
    int i = idx / NB, j = idx % NB;
    if (j <= i) L[(size_t)(base + i) * ld + base + j] = s[idx];
  }
}

// ---------- panel TRSM: L21 = A21 * L11^-T ----------
__global__ void trsm_panel(float* L, int ld, int jb) {
  extern __shared__ char smem[];
  float* s11 = (float*)smem;            // 128*128
  float* rb  = s11 + NB * NB;           // 128*(128+1)
  const int tid = threadIdx.x;          // 128
  const int base = jb * NB;
  const int r = (jb + 1) * NB + blockIdx.x * NB + tid;
  for (int idx = tid; idx < NB * NB; idx += 128)
    s11[idx] = L[(size_t)(base + idx / NB) * ld + base + idx % NB];
  float* x = rb + tid * (NB + 1);
  for (int j = 0; j < NB; j++) x[j] = L[(size_t)r * ld + base + j];
  __syncthreads();
  for (int j = 0; j < NB; j++) {
    float sum = x[j];
    for (int p = 0; p < j; p++) sum -= x[p] * s11[j * NB + p];
    x[j] = sum / s11[j * NB + j];
  }
  for (int j = 0; j < NB; j++) L[(size_t)r * ld + base + j] = x[j];
}

// ---------- triangular solves ----------
__global__ void trsv_diag_f(const float* L, int ld, float* x, int jb) {
  extern __shared__ char smem[];
  float* s = (float*)smem; float* xs = s + NB * NB;
  const int tid = threadIdx.x;  // 128
  const int base = jb * NB;
  for (int idx = tid; idx < NB * NB; idx += 128)
    s[idx] = L[(size_t)(base + idx / NB) * ld + base + idx % NB];
  xs[tid] = x[base + tid];
  __syncthreads();
  if (tid == 0) {
    for (int j = 0; j < NB; j++) {
      float v = xs[j];
      for (int p = 0; p < j; p++) v -= s[j * NB + p] * xs[p];
      xs[j] = v / s[j * NB + j];
    }
  }
  __syncthreads();
  x[base + tid] = xs[tid];
}

__global__ void trsv_upd_f(const float* __restrict__ L, int ld, float* __restrict__ x, int jb) {
  const int lane = threadIdx.x & 31;
  const int w = (blockIdx.x * blockDim.x + threadIdx.x) >> 5;  // wave per row
  const int row = (jb + 1) * NB + w;
  if (row >= N0) return;
  const int base = jb * NB;
  float sum = 0.f;
#pragma unroll
  for (int i = 0; i < 4; i++) {
    int j = lane + i * 32;
    sum += L[(size_t)row * ld + base + j] * x[base + j];
  }
  sum = wredsum(sum);
  if (lane == 0) x[row] -= sum;
}

__global__ void trsv_diag_b(const float* L, int ld, float* x, int jb) {
  extern __shared__ char smem[];
  float* s = (float*)smem; float* xs = s + NB * NB;
  const int tid = threadIdx.x;  // 128
  const int base = jb * NB;
  for (int idx = tid; idx < NB * NB; idx += 128)
    s[idx] = L[(size_t)(base + idx / NB) * ld + base + idx % NB];
  xs[tid] = x[base + tid];
  __syncthreads();
  if (tid == 0) {
    for (int i = NB - 1; i >= 0; i--) {
      float v = xs[i];
      for (int j = i + 1; j < NB; j++) v -= s[j * NB + i] * xs[j];
      xs[i] = v / s[i * NB + i];
    }
  }
  __syncthreads();
  x[base + tid] = xs[tid];
}

__global__ void trsv_upd_b(const float* __restrict__ L, int ld, float* __restrict__ x, int jb) {
  __shared__ float xs[NB];
  const int tid = threadIdx.x;  // 256
  const int base = jb * NB;
  if (tid < NB) xs[tid] = x[base + tid];
  __syncthreads();
  const int i = blockIdx.x * blockDim.x + tid;
  if (i >= base) return;
  float sum = 0.f;
  for (int j = 0; j < NB; j++) sum += L[(size_t)(base + j) * ld + i] * xs[j];
  x[i] -= sum;
}

// ---------- residual GEMV: r = A*v - bh ; rr += ||b2.*r||^2 ----------
__global__ void gemv_res(const float* __restrict__ A, const float* __restrict__ v,
                         const float* __restrict__ bh, const float* __restrict__ b2v,
                         float* __restrict__ r, float* __restrict__ rr) {
  const int lane = threadIdx.x & 31;
  const int row = (blockIdx.x * blockDim.x + threadIdx.x) >> 5;
  if (row >= N0) return;
  float sum = 0.f;
  for (int k = lane; k < N0; k += 32) sum += A[(size_t)row * N0 + k] * v[k];
  sum = wredsum(sum);
  if (lane == 0) {
    float res = sum - bh[row];
    r[row] = res;
    float tr = b2v[row] * res;
    atomicAdd(rr, tr * tr);
  }
}

__global__ void zero_rr(float* rr) { if (threadIdx.x == 0 && blockIdx.x == 0) *rr = 0.f; }

__global__ void refine_update(float* v, const float* du, const float* rr) {
  int i = blockIdx.x * blockDim.x + threadIdx.x;
  if (i >= N0) return;
  if (sqrtf(*rr) >= TOL) v[i] -= du[i];
}

__global__ void finalize(const float* v, const float* dd2v, float* out) {
  int i = blockIdx.x * blockDim.x + threadIdx.x;
  if (i < N0) out[i] = v[i] / dd2v[i];
}

// ---------- host ----------
extern "C" void kernel_launch(void* const* d_in, const int* in_sizes, int n_in,
                              void* d_out, int out_size, void* d_ws, size_t ws_size,
                              hipStream_t stream) {
  (void)in_sizes; (void)n_in; (void)out_size; (void)ws_size;
  const float* f  = (const float*)d_in[1];
  const float* d1 = (const float*)d_in[2];
  const float* B1 = (const float*)d_in[4];
  const float* B2 = (const float*)d_in[5];
  const float* D1 = (const float*)d_in[6];
  const float* D2 = (const float*)d_in[7];
  float* out = (float*)d_out;

  float* ws   = (float*)d_ws;
  float* Amat = ws;                                       // 4096x4096 A~ (pristine)
  float* Lmat = Amat + (size_t)N0 * N0;                   // 4096x4096 Cholesky factor
  unsigned short* Shi = (unsigned short*)(Lmat + (size_t)N0 * N0);  // bf16-hi of d1*sqrt(w)
  unsigned short* Slo = Shi + (size_t)N0 * N1;                      // bf16-lo
  float* w    = (float*)(Slo + (size_t)N0 * N1);          // 16384
  float* bh   = w + N1;                                   // 4096
  float* vsol = bh + N0;
  float* rvec = vsol + N0;
  float* dAdd = rvec + N0;
  float* b2v  = dAdd + N0;
  float* dd2v = b2v + N0;
  float* rr   = dd2v + N0;                                // 1

  prep_kernel<<<(N1 + 255) / 256, 256, 0, stream>>>(f, B1, B2, D1, D2, w, bh, dAdd, b2v, dd2v, rr);

  // S = d1 * sqrt(w), split to bf16 hi/lo (one streaming pass)
  {
    size_t nf4 = (size_t)N0 * N1 / 4;
    split_kernel<<<(unsigned)((nf4 + 255) / 256), 256, 0, stream>>>(d1, w, Shi, Slo);
  }

  // A~ = S S^T + diag(ridge/(b2*dd2))  [SPD], TDM-fed WMMA GEMM
  gram_gemm_tdm<<<dim3(N0 / NB, N0 / NB), 256, 2 * 4 * TILE_US * 2, stream>>>(Shi, Slo, Amat, dAdd);

  hipMemcpyAsync(Lmat, Amat, (size_t)N0 * N0 * sizeof(float), hipMemcpyDeviceToDevice, stream);

  // blocked Cholesky of A~ (lower); trailing syrk on WMMA
  for (int pb = 0; pb < N0 / NB; ++pb) {
    potf2_kernel<<<1, 256, NB * NB * 4, stream>>>(Lmat, N0, pb);
    int rem = N0 - (pb + 1) * NB;
    if (rem > 0) {
      trsm_panel<<<rem / NB, 128, (NB * NB + NB * (NB + 1)) * 4, stream>>>(Lmat, N0, pb);
      int t = rem / NB;
      const float* P = Lmat + (size_t)((pb + 1) * NB) * N0 + pb * NB;
      float* Ct      = Lmat + (size_t)((pb + 1) * NB) * N0 + (pb + 1) * NB;
      gemm128_bf16x3<<<dim3(t, t), 256, 0, stream>>>(P, N0, P, N0, NB, Ct, N0, /*lowerOnly=*/1);
    }
  }

  // solve A~ v = f/b2
  hipMemcpyAsync(vsol, bh, N0 * sizeof(float), hipMemcpyDeviceToDevice, stream);
  auto tri_solve = [&](float* x) {
    for (int jb = 0; jb < N0 / NB; jb++) {
      trsv_diag_f<<<1, 128, (NB * NB + NB) * 4, stream>>>(Lmat, N0, x, jb);
      int rows = N0 - (jb + 1) * NB;
      if (rows > 0)
        trsv_upd_f<<<(rows * 32 + 255) / 256, 256, 0, stream>>>(Lmat, N0, x, jb);
    }
    for (int jb = N0 / NB - 1; jb >= 0; jb--) {
      trsv_diag_b<<<1, 128, (NB * NB + NB) * 4, stream>>>(Lmat, N0, x, jb);
      if (jb > 0)
        trsv_upd_b<<<(jb * NB + 255) / 256, 256, 0, stream>>>(Lmat, N0, x, jb);
    }
  };
  tri_solve(vsol);

  // 5 masked iterative-refinement steps (K u - f == D_b*(A~ v - bh))
  for (int it = 0; it < 5; ++it) {
    zero_rr<<<1, 1, 0, stream>>>(rr);
    gemv_res<<<(N0 * 32) / 256, 256, 0, stream>>>(Amat, vsol, bh, b2v, rvec, rr);
    tri_solve(rvec);
    refine_update<<<(N0 + 255) / 256, 256, 0, stream>>>(vsol, rvec, rr);
  }

  // u = v ./ dd2
  finalize<<<(N0 + 255) / 256, 256, 0, stream>>>(vsol, dd2v, out);
}
